// Attention_76579266888357
// MI455X (gfx1250) — matrix operand order repaired and checked by reading, not compiled, not run on hardware
//
#include <hip/hip_runtime.h>

#define Bb   4
#define Hh   16
#define Ss   1024
#define Dd   64
#define BHn  (Bb*Hh)
#define INV_T 0.125f
#define EPSn  1e-12f

typedef __attribute__((ext_vector_type(16))) __bf16 v16bf;
typedef __attribute__((ext_vector_type(8)))  float  v8f;
typedef __attribute__((ext_vector_type(2)))  __bf16 bf16x2;

// ---- monotone float<->uint encoding for unsigned atomicMin over floats ----
static __device__ __forceinline__ unsigned enc_f(float f) {
  unsigned u = __float_as_uint(f);
  return (u & 0x80000000u) ? ~u : (u | 0x80000000u);
}
static __device__ __forceinline__ float dec_f(unsigned u) {
  return (u & 0x80000000u) ? __uint_as_float(u & 0x7fffffffu)
                           : __uint_as_float(~u);
}

// ---- A-matrix (16x32 bf16) loader: q rows, both 32-wide K slices of D=64 ----
// Layout (ISA 7.12.2): lane r=lane&15 holds row M=r; VGPR i (pair 2i,2i+1):
//   K = (i/4)*16 + (lane>=16)*8 + (i%4)*2
static __device__ __forceinline__ void load_a_qk(const float* qrow, int hi,
                                                 v16bf& a0, v16bf& a1) {
  const float2* q2 = reinterpret_cast<const float2*>(qrow);
#pragma unroll
  for (int i = 0; i < 8; ++i) {
    const int p = ((i >> 2) << 3) + (hi << 2) + (i & 3);
    float2 f0 = q2[p];       // d-slice 0..31
    float2 f1 = q2[p + 16];  // d-slice 32..63
    a0[2*i]   = (__bf16)(f0.x * INV_T);
    a0[2*i+1] = (__bf16)(f0.y * INV_T);
    a1[2*i]   = (__bf16)(f1.x * INV_T);
    a1[2*i+1] = (__bf16)(f1.y * INV_T);
  }
}

// ---- B-matrix (32x16 bf16) loader for QK^T (NT): B[K][N] = k[jb+N][dbase+K]
// Layout: lane n=lane&15 is column N; VGPR i holds K = (i/4)*16 + (lane>=16)*8 + (i%4)*2, +1
static __device__ __forceinline__ void load_b_k(const float* krow, int hi,
                                                v16bf& b0, v16bf& b1) {
  const float2* k2 = reinterpret_cast<const float2*>(krow);
#pragma unroll
  for (int i = 0; i < 8; ++i) {
    const int p = ((i >> 2) << 3) + (hi << 2) + (i & 3);
    float2 f0 = k2[p];
    float2 f1 = k2[p + 16];
    b0[2*i]   = (__bf16)f0.x;  b0[2*i+1] = (__bf16)f0.y;
    b1[2*i]   = (__bf16)f1.x;  b1[2*i+1] = (__bf16)f1.y;
  }
}

// =====================================================================
// Pass 1: global min of (q/T)·k^T via bf16 WMMA, atomicMin into ws[0]
// grid = (64 q-tiles, 64 bh), block = 256 (8 waves)
// =====================================================================
__global__ void __launch_bounds__(256, 2)
qk_min_kernel(const float* __restrict__ q, const float* __restrict__ k,
              unsigned* __restrict__ gmin) {
  const int qt = blockIdx.x, bh = blockIdx.y;
  const int tid = threadIdx.x, wave = tid >> 5, lane = tid & 31;
  const int r = lane & 15, hi = lane >> 4;
  const int q0 = qt << 4;

  __shared__ float s_wmin[8];

  v16bf a0, a1;
  load_a_qk(q + ((size_t)bh * Ss + q0 + r) * Dd, hi, a0, a1);

  float m = 3.0e38f;
  for (int t = 0; t < 8; ++t) {
    const int jb = ((wave << 3) + t) << 4;
    const float* krow = k + ((size_t)bh * Ss + jb + r) * Dd;
    if (t < 7) __builtin_prefetch(krow + 16 * Dd, 0, 0);
    v16bf b0, b1;
    load_b_k(krow, hi, b0, b1);
    v8f c = {0.f, 0.f, 0.f, 0.f, 0.f, 0.f, 0.f, 0.f};
    c = __builtin_amdgcn_wmma_f32_16x16x32_bf16(false, a0, false, b0,
                                                (short)0, c, false, false);
    c = __builtin_amdgcn_wmma_f32_16x16x32_bf16(false, a1, false, b1,
                                                (short)0, c, false, false);
#pragma unroll
    for (int i = 0; i < 8; ++i) m = fminf(m, c[i]);
  }
#pragma unroll
  for (int off = 16; off > 0; off >>= 1) m = fminf(m, __shfl_xor(m, off, 32));
  if (lane == 0) s_wmin[wave] = m;
  __syncthreads();
  if (tid == 0) {
    float mm = s_wmin[0];
#pragma unroll
    for (int w = 1; w < 8; ++w) mm = fminf(mm, s_wmin[w]);
    atomicMin(gmin, enc_f(mm));
  }
}

// =====================================================================
// Pass 2: fused scores -> scale/L2-norm/softmax -> attn store -> P·V
// grid = (64 q-tiles, 64 bh), block = 256 (8 waves)
// dynamic LDS: 64KB fp32 scores (reused for PV reduce) + 32KB bf16 P + stats
// =====================================================================
__global__ void __launch_bounds__(256, 1)
attn_fused_kernel(const float* __restrict__ q, const float* __restrict__ k,
                  const float* __restrict__ v,
                  const unsigned* __restrict__ gmin,
                  float* __restrict__ out) {
  extern __shared__ char smem[];
  float*  s_sc   = reinterpret_cast<float*>(smem);                  // [16][1024]
  __bf16* s_p    = reinterpret_cast<__bf16*>(smem + 65536);         // [16][1024]
  float*  s_inv  = reinterpret_cast<float*>(smem + 65536 + 32768);  // [16]
  float*  s_isum = s_inv + 16;                                      // [16]

  const int qt = blockIdx.x, bh = blockIdx.y;
  const int tid = threadIdx.x, wave = tid >> 5, lane = tid & 31;
  const int r = lane & 15, hi = lane >> 4;
  const int q0 = qt << 4;

  // ---- scores tile row-block into LDS ----
  v16bf a0, a1;
  load_a_qk(q + ((size_t)bh * Ss + q0 + r) * Dd, hi, a0, a1);

  for (int t = 0; t < 8; ++t) {
    const int jb = ((wave << 3) + t) << 4;
    const float* krow = k + ((size_t)bh * Ss + jb + r) * Dd;
    if (t < 7) __builtin_prefetch(krow + 16 * Dd, 0, 0);
    v16bf b0, b1;
    load_b_k(krow, hi, b0, b1);
    v8f c = {0.f, 0.f, 0.f, 0.f, 0.f, 0.f, 0.f, 0.f};
    c = __builtin_amdgcn_wmma_f32_16x16x32_bf16(false, a0, false, b0,
                                                (short)0, c, false, false);
    c = __builtin_amdgcn_wmma_f32_16x16x32_bf16(false, a1, false, b1,
                                                (short)0, c, false, false);
#pragma unroll
    for (int i = 0; i < 8; ++i)
      s_sc[(i + (hi << 3)) * Ss + jb + r] = c[i];
  }
  __syncthreads();

  const float g = fabsf(dec_f(*gmin));

  // ---- row sum of squares; fused scaling:  s / max(||s||, g*eps) ----
  {
    const int rr = tid >> 4, t16 = tid & 15;
    float ss = 0.f;
    for (int c0 = t16; c0 < Ss; c0 += 16) {
      float x = s_sc[rr * Ss + c0];
      ss += x * x;
    }
#pragma unroll
    for (int off = 8; off > 0; off >>= 1) ss += __shfl_xor(ss, off, 16);
    if (t16 == 0) s_inv[rr] = 1.0f / fmaxf(sqrtf(ss), g * EPSn);
  }
  __syncthreads();

  // ---- exp + row sum (post-norm values are in [-1,1]: no max needed) ----
  {
    const int rr = tid >> 4, t16 = tid & 15;
    const float inv = s_inv[rr];
    float psum = 0.f;
    for (int c0 = t16; c0 < Ss; c0 += 16) {
      float p = __expf(s_sc[rr * Ss + c0] * inv);
      s_sc[rr * Ss + c0] = p;
      s_p[rr * Ss + c0]  = (__bf16)p;
      psum += p;
    }
#pragma unroll
    for (int off = 8; off > 0; off >>= 1) psum += __shfl_xor(psum, off, 16);
    if (t16 == 0) s_isum[rr] = 1.0f / psum;
  }
  __syncthreads();

  // ---- store normalized attn (fp32, coalesced float4) ----
  {
    float* attn = out + (size_t)BHn * Ss * Dd + ((size_t)bh * Ss + q0) * Ss;
    for (int idx = tid; idx < 16 * 256; idx += 256) {
      const int rr = idx >> 8, c4 = (idx & 255) << 2;
      float4 pv = *reinterpret_cast<float4*>(&s_sc[rr * Ss + c4]);
      const float is = s_isum[rr];
      pv.x *= is; pv.y *= is; pv.z *= is; pv.w *= is;
      *reinterpret_cast<float4*>(attn + (size_t)rr * Ss + c4) = pv;
    }
  }
  __syncthreads();  // s_sc is reused below as the PV reduction buffer

  // ---- P·V : [16 x 1024] bf16  x  [1024 x 64] -> [16 x 64] fp32 ----
  v8f acc[4];
#pragma unroll
  for (int nt = 0; nt < 4; ++nt)
#pragma unroll
    for (int i = 0; i < 8; ++i) acc[nt][i] = 0.f;

  for (int ks = 0; ks < 4; ++ks) {
    const int kbase = ((ks << 3) + wave) << 5;  // 32-wide K slice
    v16bf pa;
#pragma unroll
    for (int i = 0; i < 8; ++i) {
      const int kk = kbase + ((i >> 2) << 4) + (hi << 3) + ((i & 3) << 1);
      bf16x2 pr = *reinterpret_cast<const bf16x2*>(&s_p[r * Ss + kk]);
      pa[2*i] = pr.x;  pa[2*i+1] = pr.y;
    }
#pragma unroll
    for (int nt = 0; nt < 4; ++nt) {
      const float* vp = v + ((size_t)bh * Ss + kbase) * Dd + (nt << 4) + r;
      v16bf vb;
#pragma unroll
      for (int i = 0; i < 8; ++i) {
        const int krow = ((i >> 2) << 4) + (hi << 3) + ((i & 3) << 1);
        vb[2*i]   = (__bf16)vp[(size_t)krow * Dd];
        vb[2*i+1] = (__bf16)vp[(size_t)(krow + 1) * Dd];
      }
      acc[nt] = __builtin_amdgcn_wmma_f32_16x16x32_bf16(
          false, pa, false, vb, (short)0, acc[nt], false, false);
    }
  }

  // cross-wave reduce in LDS (reuse s_sc), scale by 1/rowsum, store out
  float* s_red = s_sc;  // [8 waves][4 nt][8 vgpr][32 lanes] = 32KB
#pragma unroll
  for (int nt = 0; nt < 4; ++nt)
#pragma unroll
    for (int i = 0; i < 8; ++i)
      s_red[(((wave << 2) + nt) * 8 + i) * 32 + lane] = acc[nt][i];
  __syncthreads();

  for (int u = tid; u < 1024; u += 256) {
    const int nt = u >> 8, rem = u & 255, i = rem >> 5, ln = rem & 31;
    float s = 0.f;
#pragma unroll
    for (int w = 0; w < 8; ++w)
      s += s_red[(((w << 2) + nt) * 8 + i) * 32 + ln];
    const int M = i + ((ln >> 4) << 3);
    const int N = (nt << 4) + (ln & 15);
    out[((size_t)bh * Ss + q0 + M) * Dd + N] = s * s_isum[M];
  }
}

// =====================================================================
__global__ void init_min_kernel(unsigned* g) { *g = 0xFFFFFFFFu; }

__global__ void copy_v_kernel(const float4* __restrict__ src,
                              float4* __restrict__ dst, int n4) {
  const int i = blockIdx.x * blockDim.x + threadIdx.x;
  if (i < n4) dst[i] = src[i];
}

// =====================================================================
extern "C" void kernel_launch(void* const* d_in, const int* in_sizes, int n_in,
                              void* d_out, int out_size, void* d_ws,
                              size_t ws_size, hipStream_t stream) {
  (void)in_sizes; (void)n_in; (void)out_size; (void)ws_size;
  const float* q = (const float*)d_in[0];
  const float* k = (const float*)d_in[1];
  const float* v = (const float*)d_in[2];
  float* out = (float*)d_out;
  unsigned* gmin = (unsigned*)d_ws;

  init_min_kernel<<<1, 1, 0, stream>>>(gmin);

  dim3 grid(Ss / 16, BHn);
  qk_min_kernel<<<grid, 256, 0, stream>>>(q, k, gmin);

  const size_t smem = 65536 + 32768 + 256;  // scores + bf16 P + row stats
  attn_fused_kernel<<<grid, 256, smem, stream>>>(q, k, v, gmin, out);

  const int n4 = BHn * Ss * Dd / 4;
  float4* vdst = (float4*)(out + (size_t)BHn * Ss * Dd * 1 +
                           (size_t)BHn * Ss * Ss);
  copy_v_kernel<<<(n4 + 255) / 256, 256, 0, stream>>>((const float4*)v, vdst, n4);
}
